// DynamicSparseAttention_64896955843180
// MI455X (gfx1250) — matrix operand order repaired and checked by reading, not compiled, not run on hardware
//
#include <hip/hip_runtime.h>
#include <hip/hip_bf16.h>

// ---------------- problem constants ----------------
#define BATCH 4
#define SEQ   2048
#define CH    256
#define HEADS 8
#define HDIM  32          // CH / HEADS; == WMMA K for fp16 (16x16x32)
#define KEEP  614         // int(2048 * 0.3)

typedef __attribute__((ext_vector_type(16))) _Float16 v16h;
typedef __attribute__((ext_vector_type(8)))  float    v8f;
typedef unsigned int u32x4 __attribute__((ext_vector_type(4)));
typedef int          i32x8 __attribute__((ext_vector_type(8)));
typedef int          i32x4 __attribute__((ext_vector_type(4)));

#if defined(__gfx1250__) && __has_builtin(__builtin_amdgcn_tensor_load_to_lds)
#define HAS_TDM 1
#else
#define HAS_TDM 0
#endif

static __device__ __forceinline__ v8f wmma_f16(v16h a, v16h b, v8f c) {
    return __builtin_amdgcn_wmma_f32_16x16x32_f16(
        /*neg_a=*/false, a, /*neg_b=*/false, b,
        /*c_mod=*/(short)0, c, /*reuse_a=*/false, /*reuse_b=*/false);
}

// A-matrix (16x32 fp16) per-lane K index for element e (lane half = lane>=16)
static __device__ __forceinline__ int a_kidx(int e, int half) {
    return ((e & 8) ? 16 : 0) + half * 8 + (e & 7);
}
// B-matrix (32x16 fp16) per-lane K index for element e
static __device__ __forceinline__ int b_kidx(int e, int half) {
    return ((e & 8) ? 16 : 0) + half * 8 + (e & 7);
}

// order-preserving float<->uint key mapping (ascending float == ascending key)
static __device__ __forceinline__ unsigned int f2key(float f) {
    unsigned int u = __float_as_uint(f);
    return u ^ ((u >> 31) ? 0xFFFFFFFFu : 0x80000000u);
}
static __device__ __forceinline__ float key2f(unsigned int k) {
    unsigned int u = (k & 0x80000000u) ? (k ^ 0x80000000u) : ~k;
    return __uint_as_float(u);
}

// =====================================================================
// Kernel 1: qkv = x @ w_qkv^T + b_qkv   -> fp16 [s][b][h][n][d]
// M = B*N = 8192, Nout = 768, K = 256. One 16x16 tile per wave.
// =====================================================================
__global__ void qkv_gemm_kernel(const float* __restrict__ x,
                                const float* __restrict__ w,
                                const float* __restrict__ bias,
                                _Float16* __restrict__ qkv) {
    int wid  = (int)((blockIdx.x * blockDim.x + threadIdx.x) >> 5);
    int lane = (int)(threadIdx.x & 31);
    int tm = wid / 48;          // 512 row tiles
    int tn = wid % 48;          // 48 col tiles (768/16)
    int half = lane >> 4, lm = lane & 15;
    int row = tm * 16 + lm;     // A row
    int col = tn * 16 + lm;     // B col / C col

    v8f acc = {};
    for (int kk = 0; kk < 8; ++kk) {
        int kb = kk * 32;
        v16h a, bm;
#pragma unroll
        for (int e = 0; e < 16; ++e) {
            a[e]  = (_Float16)x[row * CH + kb + a_kidx(e, half)];
            bm[e] = (_Float16)w[col * CH + kb + b_kidx(e, half)];
        }
        acc = wmma_f16(a, bm, acc);
    }
    float bj = bias[col];
    int s = col >> 8;           // 0:q 1:k 2:v
    int rem = col & 255;
    int h = rem >> 5, d = rem & 31;
#pragma unroll
    for (int r = 0; r < 8; ++r) {
        int m  = tm * 16 + r + 8 * half;   // C/D row
        int bb = m >> 11, n = m & 2047;
        qkv[((((size_t)s * BATCH + bb) * HEADS + h) * SEQ + n) * HDIM + d] =
            (_Float16)(acc[r] + bj);
    }
}

// =====================================================================
// Kernel 2: attention with exact top-k mask + softmax, per (b,h,16 rows)
// LDS layout:
//   [0, 131328)        S: fp32 16 rows x 2052   (P fp16 reuses row heads)
//   [131328, 148736)   PO: per-wave partial out 8 x 16 x 34 fp32
//   [148736, 279808)   V panel staging: 2048 x 32 fp16 (TDM target)
// =====================================================================
#define S_STRIDE   2052
#define S_ROWBYTES (S_STRIDE * 4)
#define PO_STRIDE  34
#define SMEM_PO_OFF 131328
#define SMEM_V_OFF  (SMEM_PO_OFF + 8 * 16 * PO_STRIDE * 4)   // 148736
#define SMEM_TOTAL  (SMEM_V_OFF + SEQ * HDIM * 2)            // 279808

__global__ void attn_kernel(const _Float16* __restrict__ qkv,
                            float* __restrict__ attnout) {
    extern __shared__ char smem[];
    float* S  = (float*)smem;
    float* PO = (float*)(smem + SMEM_PO_OFF);
    _Float16* Vl = (_Float16*)(smem + SMEM_V_OFF);

    int blk = (int)blockIdx.x;          // B*H*(SEQ/16) = 4096
    int qt = blk & 127;
    int bh = blk >> 7;
    int h = bh & 7, b = bh >> 3;
    int tid = (int)threadIdx.x, w = tid >> 5, lane = tid & 31;
    int half = lane >> 4, lm = lane & 15;

    const _Float16* qb = qkv + (((size_t)(0 * BATCH + b) * HEADS + h) * SEQ) * HDIM;
    const _Float16* kb = qkv + (((size_t)(1 * BATCH + b) * HEADS + h) * SEQ) * HDIM;
    const _Float16* vb = qkv + (((size_t)(2 * BATCH + b) * HEADS + h) * SEQ) * HDIM;
    int qbase = qt * 16;

#if HAS_TDM
    // ---- TDM: DMA the whole V panel (2048 x 32 fp16 = 128KB) into LDS,
    // overlapped with score computation + top-k. 2D descriptor, 8B units:
    // rows = 2048 keys, 8 units/row, unit stride 8 (contiguous panel).
    if (w == 0) {
        unsigned long long gaddr = (unsigned long long)(uintptr_t)vb;
        u32x4 g0 = { 1u,                                   // count=1 valid D#
                     (unsigned int)SMEM_V_OFF,             // lds_addr
                     (unsigned int)(gaddr & 0xFFFFFFFFu),  // global_addr[31:0]
                     (unsigned int)(((gaddr >> 32) & 0x01FFFFFFu) | 0x80000000u) }; // addr[56:32] | type=2
        i32x8 g1 = { (int)(3u << 16),      // data_size=8B; no multicast/flags
                     (int)(8u << 16),      // tensor_dim0 = 8 (bits 79:48 low half)
                     (int)(2048u << 16),   // tensor_dim1 = 2048 (bits 111:80 low half)
                     (int)(8u << 16),      // tile_dim0 = 8 (bits 127:112)
                     (int)2048,            // tile_dim1 = 2048 (bits 143:128)
                     (int)8,               // tensor_dim0_stride = 8 (bits 207:160)
                     0, 0 };
        i32x4 z4 = { 0, 0, 0, 0 };
        i32x8 z8 = { 0, 0, 0, 0, 0, 0, 0, 0 };
        __builtin_amdgcn_tensor_load_to_lds(g0, g1, z4, z4, z8, 0);
    }
#else
    // fallback: cooperative wide copy of the V panel into LDS
    {
        const uint4* vsrc = (const uint4*)vb;
        uint4* vdst = (uint4*)(smem + SMEM_V_OFF);
        for (int i = tid; i < (SEQ * HDIM * 2) / 16; i += 256) vdst[i] = vsrc[i];
    }
#endif

    // warm GL2 for the K panel (global_prefetch_b8)
    __builtin_prefetch(kb + (size_t)tid * 256, 0, 1);

    // ---- phase 1: scores S = scale * Q K^T (D=32 -> single WMMA K step)
    v16h aq;
#pragma unroll
    for (int e = 0; e < 16; ++e)
        aq[e] = qb[(qbase + lm) * HDIM + a_kidx(e, half)];

    const float scale = 0.17677669529663687f;  // 32^-0.5
    for (int t = 0; t < 16; ++t) {
        int kt = w * 16 + t;                    // key tile, 128 total / 8 waves
        v16h bk;
#pragma unroll
        for (int e = 0; e < 16; ++e)
            bk[e] = kb[(kt * 16 + lm) * HDIM + b_kidx(e, half)];
        v8f c = {};
        c = wmma_f16(aq, bk, c);
#pragma unroll
        for (int r = 0; r < 8; ++r)
            S[(r + 8 * half) * S_STRIDE + kt * 16 + lm] = c[r] * scale;
    }
    __syncthreads();

    // ---- phase 2: exact k-th-largest threshold (radix binary search on
    // order-preserving uint keys cached in VGPRs), then reference-matching
    // masked softmax (dropped entries become exactly 0 and stay in the sum).
    // After the key cache is filled, S rows are dead -> fp16 P overwrites
    // the head of each row's own S storage (wave-private rows, no hazard).
    for (int rr = 0; rr < 2; ++rr) {
        int row = w + 8 * rr;                   // 16 rows over 8 waves
        float* Srow = S + row * S_STRIDE;
        unsigned int rkey[64];
#pragma unroll
        for (int j = 0; j < 64; ++j)
            rkey[j] = f2key(Srow[lane + 32 * j]);

        // largest T with count(key >= T) >= KEEP  ==> key of KEEP-th largest
        unsigned int T = 0;
        for (int bit = 31; bit >= 0; --bit) {
            unsigned int cand = T | (1u << bit);
            int cnt = 0;
#pragma unroll
            for (int j = 0; j < 64; ++j) cnt += (rkey[j] >= cand) ? 1 : 0;
#pragma unroll
            for (int off = 1; off < 32; off <<= 1) cnt += __shfl_xor(cnt, off, 32);
            if (cnt >= KEEP) T = cand;
        }
        // masked values: z = kept ? s : 0 ; dropped zeros join the softmax
        float m = 0.0f;                         // >= 0 since dropped give z=0
#pragma unroll
        for (int j = 0; j < 64; ++j) {
            float z = (rkey[j] >= T) ? key2f(rkey[j]) : 0.0f;
            m = fmaxf(m, z);
        }
#pragma unroll
        for (int off = 1; off < 32; off <<= 1) m = fmaxf(m, __shfl_xor(m, off, 32));
        float dsum = 0.0f;
#pragma unroll
        for (int j = 0; j < 64; ++j) {
            float z = (rkey[j] >= T) ? key2f(rkey[j]) : 0.0f;
            float e = __expf(z - m);
            dsum += e;
            rkey[j] = __float_as_uint(e);       // stash exp in the key regs
        }
#pragma unroll
        for (int off = 1; off < 32; off <<= 1) dsum += __shfl_xor(dsum, off, 32);
        float inv = 1.0f / dsum;
        _Float16* Prow = (_Float16*)(smem + row * S_ROWBYTES);
#pragma unroll
        for (int j = 0; j < 64; ++j)
            Prow[lane + 32 * j] = (_Float16)(__uint_as_float(rkey[j]) * inv);
    }

#if HAS_TDM
    if (w == 0) __builtin_amdgcn_s_wait_tensorcnt(0);  // V panel resident
#endif
    __syncthreads();

    // ---- phase 3: out = P @ V from LDS ; wave w owns keys [w*256, w*256+256)
    v8f acc0 = {}, acc1 = {};
    for (int ks = 0; ks < 8; ++ks) {
        int keybase = w * 256 + ks * 32;
        v16h a, b0, b1;
#pragma unroll
        for (int e = 0; e < 16; ++e) {
            const _Float16* Prow = (const _Float16*)(smem + lm * S_ROWBYTES);
            a[e] = Prow[keybase + a_kidx(e, half)];
            int kv = keybase + b_kidx(e, half);
            b0[e] = Vl[kv * HDIM + lm];
            b1[e] = Vl[kv * HDIM + 16 + lm];
        }
        acc0 = wmma_f16(a, b0, acc0);
        acc1 = wmma_f16(a, b1, acc1);
    }
#pragma unroll
    for (int r = 0; r < 8; ++r) {
        int m = r + 8 * half;
        PO[(w * 16 + m) * PO_STRIDE + lm]      = acc0[r];
        PO[(w * 16 + m) * PO_STRIDE + 16 + lm] = acc1[r];
    }
    __syncthreads();
    for (int idx = tid; idx < 512; idx += 256) {
        int m = idx >> 5, d = idx & 31;
        float s = 0.0f;
#pragma unroll
        for (int ww = 0; ww < 8; ++ww) s += PO[(ww * 16 + m) * PO_STRIDE + d];
        attnout[((size_t)(b * SEQ + qbase + m)) * CH + h * HDIM + d] = s;
    }
}

// =====================================================================
// Kernel 3: out = attnout @ w_proj^T + b_proj   (M=8192, N=256, K=256)
// =====================================================================
__global__ void proj_gemm_kernel(const float* __restrict__ ain,
                                 const float* __restrict__ w,
                                 const float* __restrict__ bias,
                                 float* __restrict__ out) {
    int wid  = (int)((blockIdx.x * blockDim.x + threadIdx.x) >> 5);
    int lane = (int)(threadIdx.x & 31);
    int tm = wid >> 4;          // 512 row tiles
    int tn = wid & 15;          // 16 col tiles
    int half = lane >> 4, lm = lane & 15;
    int row = tm * 16 + lm;
    int col = tn * 16 + lm;

    v8f acc = {};
    for (int kk = 0; kk < 8; ++kk) {
        int kbase = kk * 32;
        v16h a, bm;
#pragma unroll
        for (int e = 0; e < 16; ++e) {
            a[e]  = (_Float16)ain[row * CH + kbase + a_kidx(e, half)];
            bm[e] = (_Float16)w[col * CH + kbase + b_kidx(e, half)];
        }
        acc = wmma_f16(a, bm, acc);
    }
    float bj = bias[col];
#pragma unroll
    for (int r = 0; r < 8; ++r)
        out[(size_t)(tm * 16 + r + 8 * half) * CH + col] = acc[r] + bj;
}

// =====================================================================
extern "C" void kernel_launch(void* const* d_in, const int* in_sizes, int n_in,
                              void* d_out, int out_size, void* d_ws, size_t ws_size,
                              hipStream_t stream) {
    const float* x      = (const float*)d_in[0];
    const float* w_qkv  = (const float*)d_in[1];
    const float* b_qkv  = (const float*)d_in[2];
    const float* w_proj = (const float*)d_in[3];
    const float* b_proj = (const float*)d_in[4];
    float* out = (float*)d_out;

    // workspace layout: fp16 qkv [3][B][H][N][D] then fp32 attn output
    _Float16* qkvh   = (_Float16*)d_ws;
    size_t qkv_bytes = (size_t)3 * BATCH * HEADS * SEQ * HDIM * sizeof(_Float16);
    float* attnout   = (float*)((char*)d_ws + qkv_bytes);

    (void)in_sizes; (void)n_in; (void)out_size; (void)ws_size;

    // allow > 64KB dynamic LDS for the attention kernel (host-side, capture-safe)
    (void)hipFuncSetAttribute((const void*)attn_kernel,
                              hipFuncAttributeMaxDynamicSharedMemorySize, SMEM_TOTAL);

    // 1) QKV: 512*48 = 24576 wave-tiles, 8 waves/block
    qkv_gemm_kernel<<<3072, 256, 0, stream>>>(x, w_qkv, b_qkv, qkvh);

    // 2) attention: one block per (b, h, 16-query-row tile)
    attn_kernel<<<BATCH * HEADS * (SEQ / 16), 256, SMEM_TOTAL, stream>>>(qkvh, attnout);

    // 3) projection: 512*16 = 8192 wave-tiles, 8 waves/block
    proj_gemm_kernel<<<1024, 256, 0, stream>>>(attnout, w_proj, b_proj, out);
}
